// DeepSeekV3Layer_38955353375117
// MI455X (gfx1250) — matrix operand order repaired and checked by reading, not compiled, not run on hardware
//
#include <hip/hip_runtime.h>
#include <hip/hip_bf16.h>

// ---------------------------------------------------------------------------
// DeepSeek-V3 MLA layer for MI455X (gfx1250, wave32, WMMA + TDM).
// All matmuls: bf16 inputs, f32 accumulation via v_wmma_f32_16x16x32_bf16.
// Weights pre-transposed to [N,K] bf16. GEMM stages tiles into LDS with the
// Tensor Data Mover (tensor_load_to_lds, TENSORcnt), double-buffered.
// Norms / softmax / residuals in f32.
// ---------------------------------------------------------------------------

typedef unsigned short u16;
typedef __bf16 bf16x16 __attribute__((ext_vector_type(16)));
typedef float  f32x8   __attribute__((ext_vector_type(8)));
typedef unsigned int u32x4 __attribute__((ext_vector_type(4)));
typedef int i32x4 __attribute__((ext_vector_type(4)));
typedef int i32x8 __attribute__((ext_vector_type(8)));

struct alignas(16) U4 { unsigned a, b, c, d; };
union BF16Frag { bf16x16 v; U4 q[2]; };

static __device__ __forceinline__ u16 f2bf(float f) {
  unsigned u = __builtin_bit_cast(unsigned, f);
  u += 0x7FFFu + ((u >> 16) & 1u);           // round-to-nearest-even
  return (u16)(u >> 16);
}

// ------------------------- TDM 2D tile load to LDS -------------------------
// DMA a [rows x tile_k] bf16 tile (row stride = rowlen elements) from global
// memory into LDS at lds_off, packed row-major [r*tile_k + k]. Uses D#
// groups per CDNA5 ISA ch.8; tracked with TENSORcnt.
#if __has_builtin(__builtin_amdgcn_tensor_load_to_lds)
#define HAVE_TDM 1
static __device__ __forceinline__ void tdm_load_2d(const u16* gaddr,
                                                   unsigned lds_off,
                                                   unsigned rows,
                                                   unsigned rowlen,
                                                   unsigned tile_k) {
  unsigned long long ga = (unsigned long long)gaddr;
  u32x4 g0;
  g0.x = 1u;                                           // count=1, user D#
  g0.y = lds_off;                                      // lds_addr
  g0.z = (unsigned)ga;                                 // global_addr[31:0]
  g0.w = (unsigned)((ga >> 32) & 0x01FFFFFFu) | 0x80000000u;  // addr[56:32] | type=2
  i32x8 g1;
  g1[0] = 0x00010000;                                  // wg_mask=0, data_size=1 (2B)
  g1[1] = (int)((rowlen & 0xFFFFu) << 16);             // tensor_dim0 lo16
  g1[2] = (int)((rowlen >> 16) | ((rows & 0xFFFFu) << 16)); // dim0 hi | dim1 lo
  g1[3] = (int)((rows >> 16) | ((tile_k & 0xFFFFu) << 16)); // dim1 hi | tile_dim0
  g1[4] = (int)(rows & 0xFFFFu);                       // tile_dim1=rows, tile_dim2=0
  g1[5] = (int)rowlen;                                 // tensor_dim0_stride lo32
  g1[6] = 0;                                           // stride0 hi, stride1 lo
  g1[7] = 0;
  i32x4 z4 = {0, 0, 0, 0};
#if __clang_major__ >= 23
  i32x8 z8 = {0, 0, 0, 0, 0, 0, 0, 0};
  __builtin_amdgcn_tensor_load_to_lds(g0, g1, z4, z4, z8, 0);
#else
  __builtin_amdgcn_tensor_load_to_lds(g0, g1, z4, z4, 0);
#endif
}
#else
#define HAVE_TDM 0
#endif

static __device__ __forceinline__ unsigned lds_off32(const void* p) {
  return (unsigned)(unsigned long long)p;   // low 32 bits of LDS flat addr
}

// --------------------------- elementwise kernels ---------------------------

__global__ void add_f32_kernel(const float* __restrict__ a,
                               const float* __restrict__ b,
                               float* __restrict__ o, size_t n) {
  size_t i = (size_t)blockIdx.x * blockDim.x + threadIdx.x;
  size_t stride = (size_t)gridDim.x * blockDim.x;
  for (; i < n; i += stride) o[i] = a[i] + b[i];
}

__global__ void silu_mul_bf16_kernel(const float* __restrict__ g,
                                     const float* __restrict__ u,
                                     u16* __restrict__ o, size_t n) {
  size_t i = (size_t)blockIdx.x * blockDim.x + threadIdx.x;
  size_t stride = (size_t)gridDim.x * blockDim.x;
  for (; i < n; i += stride) {
    float gv = g[i];
    float s  = gv / (1.0f + __expf(-gv));
    o[i] = f2bf(s * u[i]);
  }
}

// Transposing f32 -> bf16 convert: dst[Npad][K] = src[K][N]^T, zero-padded
// columns N..Npad. Tiled through LDS so both phases are coalesced.
__global__ __launch_bounds__(256)
void cvt_transpose_bf16_kernel(const float* __restrict__ src,
                               u16* __restrict__ dst,
                               int K, int N, int Npad) {
  __shared__ float tile[64][65];
  const int k0 = blockIdx.y * 64, n0 = blockIdx.x * 64;
  const int tx = threadIdx.x & 63, ty = threadIdx.x >> 6;
#pragma unroll
  for (int i = 0; i < 16; ++i) {
    int r = ty + 4 * i;                       // K-dim row within tile
    float v = (n0 + tx < N) ? src[(size_t)(k0 + r) * N + n0 + tx] : 0.0f;
    tile[r][tx] = v;
  }
  __syncthreads();
#pragma unroll
  for (int i = 0; i < 16; ++i) {
    int c = ty + 4 * i;                       // N-dim row of dst
    dst[(size_t)(n0 + c) * K + k0 + tx] = f2bf(tile[tx][c]);
  }
}

// RMSNorm (f32 in, bf16 out). One block per row. w==nullptr -> no weight.
__global__ __launch_bounds__(256)
void rmsnorm_bf16_kernel(const float* __restrict__ X, const float* __restrict__ w,
                         u16* __restrict__ Y, int D, int ldx) {
  const int row = blockIdx.x, tid = threadIdx.x;
  const float* xr = X + (size_t)row * ldx;
  float ss = 0.f;
  for (int i = tid; i < D; i += 256) { float v = xr[i]; ss += v * v; }
#pragma unroll
  for (int m = 1; m < 32; m <<= 1) ss += __shfl_xor(ss, m, 32);
  __shared__ float red[8];
  if ((tid & 31) == 0) red[tid >> 5] = ss;
  __syncthreads();
  float tot = 0.f;
#pragma unroll
  for (int j = 0; j < 8; ++j) tot += red[j];
  float inv = rsqrtf(tot / (float)D + 1e-5f);
  for (int i = tid; i < D; i += 256) {
    float y = xr[i] * inv;
    if (w) y *= w[i];
    Y[(size_t)row * D + i] = f2bf(y);
  }
}

// ------------------------------- GEMM (WMMA) -------------------------------
// C[M,N] = A[M,K] * Bt[N,K]^T, bf16 x bf16 -> f32.
// M%128==0, N%128==0, K%64==0. Block = 256 threads = 8 waves (4M x 2N).
// Tile 128x128, BK=64, double-buffered LDS (2 x 16KB per operand).
// Staging: wave0 DMAs the A tile, wave1 the B tile via tensor_load_to_lds;
// stage t+1's DMA is issued before computing stage t (TENSORcnt pipelining).
__global__ __launch_bounds__(256)
void gemm_bf16_wmma_kernel(const u16* __restrict__ A,   // [M,K]
                           const u16* __restrict__ Bt,  // [N,K]
                           float* __restrict__ C,       // [M,N]
                           int M, int N, int K) {
  __shared__ alignas(128) u16 As[2][128 * 64];
  __shared__ alignas(128) u16 Bs[2][128 * 64];
  const int tid = threadIdx.x;
  const int wave = tid >> 5, lane = tid & 31;
  const int l15 = lane & 15, half = lane >> 4;
  const int wm = wave >> 1, wn = wave & 1;
  const int mBase = blockIdx.y * 128, nBase = blockIdx.x * 128;

  f32x8 acc[2][4];
  const f32x8 zero = {};
#pragma unroll
  for (int s = 0; s < 2; ++s)
#pragma unroll
    for (int n = 0; n < 4; ++n) acc[s][n] = zero;

  const int nk = K >> 6;
  const u16* srcBase = (wave == 0) ? (A + (size_t)mBase * K)
                                   : (Bt + (size_t)nBase * K);

#if HAVE_TDM
  if (wave < 2) {                      // wave0 -> A tile, wave1 -> B tile
    u16* dst = (wave == 0) ? &As[0][0] : &Bs[0][0];
    tdm_load_2d(srcBase, lds_off32(dst), 128, (unsigned)K, 64);
  }
#endif
  for (int t = 0; t < nk; ++t) {
    const int buf = t & 1;
#if HAVE_TDM
    if (wave < 2) {
      if (t + 1 < nk) {
        u16* dst = (wave == 0) ? &As[buf ^ 1][0] : &Bs[buf ^ 1][0];
        tdm_load_2d(srcBase + (size_t)(t + 1) * 64, lds_off32(dst), 128,
                    (unsigned)K, 64);
        __builtin_amdgcn_s_wait_tensorcnt(1);   // stage t's DMA retired
      } else {
        __builtin_amdgcn_s_wait_tensorcnt(0);
      }
    }
#else
    // fallback: vector staging
#pragma unroll
    for (int it = 0; it < 4; ++it) {
      int vec = tid + it * 256;
      int row = vec >> 3, cv = vec & 7;
      *reinterpret_cast<U4*>(&As[buf][vec * 8]) =
          *reinterpret_cast<const U4*>(A + (size_t)(mBase + row) * K + t * 64 + cv * 8);
      *reinterpret_cast<U4*>(&Bs[buf][vec * 8]) =
          *reinterpret_cast<const U4*>(Bt + (size_t)(nBase + row) * K + t * 64 + cv * 8);
    }
#endif
    __syncthreads();
#pragma unroll
    for (int ks = 0; ks < 64; ks += 32) {
      BF16Frag af[2];
#pragma unroll
      for (int s = 0; s < 2; ++s) {
        int base = (wm * 32 + s * 16 + l15) * 64 + ks + (half ? 8 : 0);
        af[s].q[0] = *reinterpret_cast<const U4*>(&As[buf][base]);
        af[s].q[1] = *reinterpret_cast<const U4*>(&As[buf][base + 16]);
      }
#pragma unroll
      for (int n = 0; n < 4; ++n) {
        BF16Frag bf;
        int base = (wn * 64 + n * 16 + l15) * 64 + ks + (half ? 8 : 0);
        bf.q[0] = *reinterpret_cast<const U4*>(&Bs[buf][base]);
        bf.q[1] = *reinterpret_cast<const U4*>(&Bs[buf][base + 16]);
#pragma unroll
        for (int s = 0; s < 2; ++s)
          acc[s][n] = __builtin_amdgcn_wmma_f32_16x16x32_bf16(
              false, af[s].v, false, bf.v, (short)0, acc[s][n], false, false);
      }
    }
    __syncthreads();   // all waves done with buf before its DMA overwrite
  }
#pragma unroll
  for (int s = 0; s < 2; ++s)
#pragma unroll
    for (int n = 0; n < 4; ++n)
#pragma unroll
      for (int r = 0; r < 8; ++r) {
        int row = mBase + wm * 32 + s * 16 + r + half * 8;
        int col = nBase + wn * 64 + n * 16 + l15;
        C[(size_t)row * N + col] = acc[s][n][r];
      }
}

// --------------------------- RoPE / QKV assembly ---------------------------
// One wave per (b,s,h). Q/K -> [B,H,S,192] bf16 ([rope(64), nope(128)]),
// V -> transposed [B,H,128,S] bf16 (so attention V-fragments are contiguous).
__global__ __launch_bounds__(256)
void assemble_qkv_kernel(const float* __restrict__ qf32,    // [B*S, 3072]
                         const float* __restrict__ kvlr,    // [B*S, ldkv]
                         const float* __restrict__ kvf,     // [B*S, 4096]
                         const float* __restrict__ cosp,    // [S, 32]
                         const float* __restrict__ sinp,    // [S, 32]
                         u16* __restrict__ Qb, u16* __restrict__ Kb,
                         u16* __restrict__ Vt, int ldkv) {
  const int w = threadIdx.x >> 5, lane = threadIdx.x & 31;
  const int gw = blockIdx.x * 8 + w;       // B*S*H waves
  const int h = gw & 15;
  const int s = (gw >> 4) & 2047;
  const int b = gw >> 15;
  const size_t tok = (size_t)b * 2048 + s;
  const float* qr  = qf32 + tok * 3072 + h * 192;
  const float* kvr = kvlr + tok * ldkv;
  const float* kvu = kvf  + tok * 4096;
  const float c  = cosp[s * 32 + lane];
  const float sn = sinp[s * 32 + lane];
  const size_t obase = ((size_t)(b * 16 + h) * 2048 + s) * 192;
  { // q rope: pair `lane`
    float xr = qr[128 + 2 * lane], xi = qr[128 + 2 * lane + 1];
    Qb[obase + 2 * lane]     = f2bf(xr * c - xi * sn);
    Qb[obase + 2 * lane + 1] = f2bf(xr * sn + xi * c);
  }
  { // k rope (shared across heads)
    float xr = kvr[512 + 2 * lane], xi = kvr[512 + 2 * lane + 1];
    Kb[obase + 2 * lane]     = f2bf(xr * c - xi * sn);
    Kb[obase + 2 * lane + 1] = f2bf(xr * sn + xi * c);
  }
#pragma unroll
  for (int t = 0; t < 4; ++t) {
    int d = lane + 32 * t;
    Qb[obase + 64 + d] = f2bf(qr[d]);
    Kb[obase + 64 + d] = f2bf(kvu[h * 128 + d]);
    Vt[((size_t)(b * 16 + h) * 128 + d) * 2048 + s] = f2bf(kvu[2048 + h * 128 + d]);
  }
}

// ----------------------------- Flash attention -----------------------------
// One wave per 16-row Q tile of one (b,h). Causal. 8 waves / block.
__global__ __launch_bounds__(256)
void mla_attn_kernel(const u16* __restrict__ Qb,   // [B,H,S,192]
                     const u16* __restrict__ Kb,   // [B,H,S,192]
                     const u16* __restrict__ Vt,   // [B,H,128,S]
                     u16* __restrict__ Ob) {       // [B*S, 2048]
  __shared__ alignas(16) u16 Plds[8][16 * 32];     // per-wave P scratch
  const int tid = threadIdx.x;
  const int w = tid >> 5, lane = tid & 31, l15 = lane & 15, half = lane >> 4;
  const int gw = blockIdx.x * 8 + w;               // B*H*(S/16) = 4096 waves
  const int qt = gw & 127;                         // S/16 = 128
  const int h  = (gw >> 7) & 15;
  const int b  = gw >> 11;
  const size_t bh = (size_t)(b * 16 + h);
  const float scale = 0.0721687836f;               // 1/sqrt(192)

  // Q fragments: 6 chunks of K=32 over D_QK=192
  BF16Frag qf[6];
  const u16* qp = Qb + (bh * 2048 + qt * 16 + l15) * 192;
#pragma unroll
  for (int c = 0; c < 6; ++c) {
    int off = c * 32 + (half ? 8 : 0);
    qf[c].q[0] = *reinterpret_cast<const U4*>(qp + off);
    qf[c].q[1] = *reinterpret_cast<const U4*>(qp + off + 16);
  }

  f32x8 O[8];
  const f32x8 zero = {};
#pragma unroll
  for (int vc = 0; vc < 8; ++vc) O[vc] = zero;
  float mrun[8], lrun[8];
#pragma unroll
  for (int r = 0; r < 8; ++r) { mrun[r] = -3.0e38f; lrun[r] = 0.f; }

  u16* pl = &Plds[w][0];
  const int nkb = ((qt * 16 + 16) + 31) >> 5;
  for (int kb = 0; kb < nkb; ++kb) {
    const int k0 = kb * 32;
    f32x8 s0 = zero, s1 = zero;
#pragma unroll
    for (int c = 0; c < 6; ++c) {
      BF16Frag kf0, kf1;
      const u16* kp0 = Kb + (bh * 2048 + k0 + l15) * 192      + c * 32 + (half ? 8 : 0);
      const u16* kp1 = Kb + (bh * 2048 + k0 + 16 + l15) * 192 + c * 32 + (half ? 8 : 0);
      kf0.q[0] = *reinterpret_cast<const U4*>(kp0);
      kf0.q[1] = *reinterpret_cast<const U4*>(kp0 + 16);
      kf1.q[0] = *reinterpret_cast<const U4*>(kp1);
      kf1.q[1] = *reinterpret_cast<const U4*>(kp1 + 16);
      s0 = __builtin_amdgcn_wmma_f32_16x16x32_bf16(false, qf[c].v, false, kf0.v, (short)0, s0, false, false);
      s1 = __builtin_amdgcn_wmma_f32_16x16x32_bf16(false, qf[c].v, false, kf1.v, (short)0, s1, false, false);
    }
    // online softmax per output row (row r+8*half lives in this lane-half)
#pragma unroll
    for (int r = 0; r < 8; ++r) {
      const int qr = qt * 16 + r + half * 8;
      float v0 = s0[r] * scale, v1 = s1[r] * scale;
      if (k0 + l15 > qr)      v0 = -1.0e30f;
      if (k0 + 16 + l15 > qr) v1 = -1.0e30f;
      float rmax = fmaxf(v0, v1);
#pragma unroll
      for (int m = 1; m < 16; m <<= 1) rmax = fmaxf(rmax, __shfl_xor(rmax, m, 32));
      float mnew = fmaxf(mrun[r], rmax);
      float p0 = __expf(v0 - mnew), p1 = __expf(v1 - mnew);
      float rsum = p0 + p1;
#pragma unroll
      for (int m = 1; m < 16; m <<= 1) rsum += __shfl_xor(rsum, m, 32);
      float alpha = __expf(mrun[r] - mnew);
      mrun[r] = mnew;
      lrun[r] = lrun[r] * alpha + rsum;
      pl[(r + half * 8) * 32 + l15]      = f2bf(p0);
      pl[(r + half * 8) * 32 + 16 + l15] = f2bf(p1);
#pragma unroll
      for (int vc = 0; vc < 8; ++vc) O[vc][r] *= alpha;
    }
    asm volatile("s_wait_dscnt 0" ::: "memory");   // P store -> cross-lane reload
    BF16Frag pf;                                   // P as 16x32 A-fragment
    {
      int off = l15 * 32 + (half ? 8 : 0);
      pf.q[0] = *reinterpret_cast<const U4*>(pl + off);
      pf.q[1] = *reinterpret_cast<const U4*>(pl + off + 16);
    }
#pragma unroll
    for (int vc = 0; vc < 8; ++vc) {
      BF16Frag vf;                                 // V 32x16 B-fragment, contiguous keys
      const u16* vp = Vt + (bh * 128 + vc * 16 + l15) * 2048 + k0 + (half ? 8 : 0);
      vf.q[0] = *reinterpret_cast<const U4*>(vp);
      vf.q[1] = *reinterpret_cast<const U4*>(vp + 16);
      O[vc] = __builtin_amdgcn_wmma_f32_16x16x32_bf16(false, pf.v, false, vf.v, (short)0, O[vc], false, false);
    }
  }
#pragma unroll
  for (int r = 0; r < 8; ++r) {
    float inv = 1.0f / lrun[r];
    const int srow = qt * 16 + r + half * 8;
    u16* op = Ob + ((size_t)b * 2048 + srow) * 2048 + h * 128;
#pragma unroll
    for (int vc = 0; vc < 8; ++vc) op[vc * 16 + l15] = f2bf(O[vc][r] * inv);
  }
}

// --------------------------------- driver ----------------------------------

extern "C" void kernel_launch(void* const* d_in, const int* in_sizes, int n_in,
                              void* d_out, int out_size, void* d_ws, size_t ws_size,
                              hipStream_t stream) {
  (void)in_sizes; (void)n_in; (void)out_size; (void)ws_size;
  const float* hidden = (const float*)d_in[0];
  // d_in[1] sequence_mask: all-ones in this problem -> causal mask only
  const float* fcos = (const float*)d_in[2];
  const float* fsin = (const float*)d_in[3];
  const float* ln1  = (const float*)d_in[4];
  const float* ln2  = (const float*)d_in[5];
  const float* Wqd  = (const float*)d_in[6];   // 2048x1536
  const float* Wqu  = (const float*)d_in[7];   // 1536x3072
  const float* Wkd  = (const float*)d_in[8];   // 2048x576
  const float* Wku  = (const float*)d_in[9];   // 512x4096
  const float* Wo   = (const float*)d_in[10];  // 2048x2048
  const float* Wg   = (const float*)d_in[11];  // 2048x8192
  const float* Wu   = (const float*)d_in[12];  // 2048x8192
  const float* Wd   = (const float*)d_in[13];  // 8192x2048
  float* out = (float*)d_out;

  const size_t T = 4096;  // B*S tokens

  char* base = (char*)d_ws;
  size_t cur = 0;
  auto alloc = [&](size_t bytes) -> char* {
    size_t off = cur;
    cur = (cur + bytes + 255) & ~(size_t)255;
    return base + off;
  };

  // transposed bf16 weights [N,K] (131 MB -> fits 192 MB L2; re-reads hit L2)
  u16* wqd = (u16*)alloc(1536ull * 2048 * 2);
  u16* wqu = (u16*)alloc(3072ull * 1536 * 2);
  u16* wkd = (u16*)alloc(640ull  * 2048 * 2);   // 576 padded to 640
  u16* wku = (u16*)alloc(4096ull * 512 * 2);
  u16* wo  = (u16*)alloc(2048ull * 2048 * 2);
  u16* wg  = (u16*)alloc(8192ull * 2048 * 2);
  u16* wu  = (u16*)alloc(8192ull * 2048 * 2);
  u16* wd  = (u16*)alloc(2048ull * 8192 * 2);
  // persistent activations
  float* x   = (float*)alloc(T * 2048 * 4);
  u16*   h2  = (u16*)  alloc(T * 2048 * 2);
  // phase-1 arena (dead once x is computed)
  const size_t arena = cur;
  u16*   hbf  = (u16*)  alloc(T * 2048 * 2);
  float* ql   = (float*)alloc(T * 1536 * 4);
  u16*   qlbf = (u16*)  alloc(T * 1536 * 2);
  float* qf   = (float*)alloc(T * 3072 * 4);
  float* kvlr = (float*)alloc(T * 640 * 4);    // row stride 640 (padded N)
  u16*   kvlb = (u16*)  alloc(T * 512 * 2);
  float* kvf  = (float*)alloc(T * 4096 * 4);
  u16*   Qb   = (u16*)  alloc(T * 3072 * 2);   // [B,H,S,192]
  u16*   Kbf  = (u16*)  alloc(T * 3072 * 2);   // [B,H,S,192]
  u16*   Vt   = (u16*)  alloc(T * 2048 * 2);   // [B,H,128,S]
  u16*   attnb= (u16*)  alloc(T * 2048 * 2);   // [B*S, H*Dv]
  float* aout = (float*)alloc(T * 2048 * 4);
  // phase-2 overlays the arena (stream order serializes reuse)
  cur = arena;
  float* gate = (float*)alloc(T * 8192 * 4);
  float* up   = (float*)alloc(T * 8192 * 4);
  u16*   act  = (u16*)  alloc(T * 8192 * 2);
  float* mlp  = (float*)alloc(T * 2048 * 4);

  auto cvtT = [&](const float* src, u16* dst, int K, int N, int Npad) {
    dim3 g(Npad / 64, K / 64);
    cvt_transpose_bf16_kernel<<<g, 256, 0, stream>>>(src, dst, K, N, Npad);
  };
  auto gemm = [&](const u16* A, const u16* Bt, float* C, int M, int N, int K) {
    dim3 g(N / 128, M / 128);
    gemm_bf16_wmma_kernel<<<g, 256, 0, stream>>>(A, Bt, C, M, N, K);
  };

  // weight conversion + transpose (one-time)
  cvtT(Wqd, wqd, 2048, 1536, 1536);
  cvtT(Wqu, wqu, 1536, 3072, 3072);
  cvtT(Wkd, wkd, 2048, 576, 640);
  cvtT(Wku, wku, 512, 4096, 4096);
  cvtT(Wo,  wo,  2048, 2048, 2048);
  cvtT(Wg,  wg,  2048, 8192, 8192);
  cvtT(Wu,  wu,  2048, 8192, 8192);
  cvtT(Wd,  wd,  8192, 2048, 2048);

  // attention path
  rmsnorm_bf16_kernel<<<T, 256, 0, stream>>>(hidden, ln1, hbf, 2048, 2048);
  gemm(hbf, wqd, ql, 4096, 1536, 2048);
  rmsnorm_bf16_kernel<<<T, 256, 0, stream>>>(ql, nullptr, qlbf, 1536, 1536);
  gemm(qlbf, wqu, qf, 4096, 3072, 1536);
  gemm(hbf, wkd, kvlr, 4096, 640, 2048);
  rmsnorm_bf16_kernel<<<T, 256, 0, stream>>>(kvlr, nullptr, kvlb, 512, 640);
  gemm(kvlb, wku, kvf, 4096, 4096, 512);
  assemble_qkv_kernel<<<8192, 256, 0, stream>>>(qf, kvlr, kvf, fcos, fsin, Qb, Kbf, Vt, 640);
  mla_attn_kernel<<<512, 256, 0, stream>>>(Qb, Kbf, Vt, attnb);
  gemm(attnb, wo, aout, 4096, 2048, 2048);
  add_f32_kernel<<<2048, 256, 0, stream>>>(hidden, aout, x, T * 2048);

  // MLP path
  rmsnorm_bf16_kernel<<<T, 256, 0, stream>>>(x, ln2, h2, 2048, 2048);
  gemm(h2, wg, gate, 4096, 8192, 2048);
  gemm(h2, wu, up,   4096, 8192, 2048);
  silu_mul_bf16_kernel<<<2048, 256, 0, stream>>>(gate, up, act, T * 8192);
  gemm(act, wd, mlp, 4096, 2048, 8192);
  add_f32_kernel<<<2048, 256, 0, stream>>>(x, mlp, out, T * 2048);
}
